// BidirectionalRNN_86595130622324
// MI455X (gfx1250) — matrix operand order repaired and checked by reading, not compiled, not run on hardware
//
#include <hip/hip_runtime.h>
#include <hip/hip_fp16.h>

typedef __attribute__((ext_vector_type(16))) _Float16 v16h;
typedef __attribute__((ext_vector_type(8)))  _Float16 v8h;
typedef __attribute__((ext_vector_type(8)))  float    v8f;

constexpr int cB = 32, cT = 512, cD = 64, cH = 128, cE = 256, cNH = 8, cHD = 32;
constexpr int cTB = cT * cB;   // 16384 rows in the (t,b)-flattened matrices

#define WMMA_F16(a, b, c) \
  __builtin_amdgcn_wmma_f32_16x16x32_f16(false, (a), false, (b), (short)0, (c), false, false)

// ---- fast transcendentals (v_rcp_f32 / v_exp_f32, no IEEE div sequences) --

__device__ __forceinline__ float fast_rcp(float x) { return __builtin_amdgcn_rcpf(x); }
__device__ __forceinline__ float fast_sigmoid(float g) {
  return fast_rcp(1.f + __expf(-g));
}
__device__ __forceinline__ float fast_tanh(float g) {
  // tanh(g) = 1 - 2/(exp(2g)+1); saturates to +-1 correctly
  return 1.f - 2.f * fast_rcp(1.f + __expf(2.f * g));
}

// ---- fragment loaders (layouts per CDNA5 ISA §7.12.2) ----------

// A 16x32 f16: lane l holds row (row0 + (l&15)); K halves {kb..kb+7, kb+16..kb+23}, kb = 8*(l>=16)
__device__ __forceinline__ v16h frag_from_split(const _Float16* p) {
  v8h lo = *(const v8h*)(p);
  v8h hi = *(const v8h*)(p + 16);
  v16h f;
#pragma unroll
  for (int e = 0; e < 8; ++e) { f[e] = lo[e]; f[e + 8] = hi[e]; }
  return f;
}
// 16 contiguous K halves from p; unused, B fragments (W rows) are loaded split like A via frag_b
__device__ __forceinline__ v16h frag_from_contig(const _Float16* p) {
  v8h lo = *(const v8h*)(p);
  v8h hi = *(const v8h*)(p + 8);
  v16h f;
#pragma unroll
  for (int e = 0; e < 8; ++e) { f[e] = lo[e]; f[e + 8] = hi[e]; }
  return f;
}
__device__ __forceinline__ v16h frag_a(const _Float16* base, int ld, int row0, int k0) {
  const int lane = threadIdx.x & 31;
  return frag_from_split(base + (row0 + (lane & 15)) * ld + k0 + ((lane & 16) ? 8 : 0));
}
__device__ __forceinline__ v16h frag_b(const _Float16* base, int ld, int col0, int k0) {
  const int lane = threadIdx.x & 31;
  return frag_from_split(base + (col0 + (lane & 15)) * ld + k0 + ((lane & 16) ? 8 : 0));
}

// ---- small elementwise kernels -------------------------------------------

__global__ void k_prep_xm(const float* __restrict__ x, const float* __restrict__ mask,
                          _Float16* __restrict__ xm) {
  int i = blockIdx.x * blockDim.x + threadIdx.x;
  if (i >= cTB * cD) return;
  int d = i % cD, r = i / cD;
  int t = r / cB, b = r % cB;
  int src = (b * cT + t) * cD + d;
  xm[i] = (_Float16)(x[src] * mask[src]);
}

__global__ void k_cvt(const float* __restrict__ x, _Float16* __restrict__ y, int n) {
  int i = blockIdx.x * blockDim.x + threadIdx.x;
  if (i < n) y[i] = (_Float16)x[i];
}

// vT[(b*NH+h)*HD + d][t] = V[t,b,h,d]   (V = qkv cols [2E, 3E))
__global__ void k_transpose_v(const _Float16* __restrict__ qkv, _Float16* __restrict__ vT) {
  int i = blockIdx.x * blockDim.x + threadIdx.x;
  if (i >= cB * cNH * cHD * cT) return;
  int t = i & (cT - 1);
  int d = (i >> 9) & (cHD - 1);
  int bh = i >> 14;
  int b = bh >> 3, hh = bh & 7;
  vT[i] = qkv[(t * cB + b) * (3 * cE) + 2 * cE + hh * cHD + d];
}

__global__ void k_final(const float* __restrict__ x, const float* __restrict__ mask,
                        const float* __restrict__ imp, float* __restrict__ out) {
  int i = blockIdx.x * blockDim.x + threadIdx.x;
  if (i >= cB * cT * cD) return;
  int d = i % cD, t = (i / cD) % cT, b = i / (cD * cT);
  float iv = imp[(t * cB + b) * cD + d];
  float mv = mask[i];
  out[i] = x[i] * mv + iv * (1.f - mv);
  out[cB * cT * cD + i] = iv;  // second tuple output: imputed [B,T,D]
}

// ---- generic WMMA GEMM: Y[M,N] = A[M,K] * W[N,K]^T + bias[N] --------------
// block = 4 waves, wave w computes 16x16 tile (blockIdx.x, blockIdx.y*4 + w)
// K templated -> fully unrolled K chain, clause-friendly loads.

template <int K, bool OUT16>
__global__ void __launch_bounds__(128)
gemm_xwT(const _Float16* __restrict__ A, const _Float16* __restrict__ W,
         const float* __restrict__ bias, float* __restrict__ C32,
         _Float16* __restrict__ C16, int N) {
  const int mt = blockIdx.x;
  const int nt = blockIdx.y * 4 + (threadIdx.x >> 5);
  const int lane = threadIdx.x & 31;
  v8f acc = {};
#pragma unroll
  for (int k = 0; k < K; k += 32) {
    v16h a = frag_a(A, K, mt * 16, k);
    v16h b = frag_b(W, K, nt * 16, k);
    acc = WMMA_F16(a, b, acc);
  }
  const int col = nt * 16 + (lane & 15);
  const float bv = bias[col];
  const int rbase = mt * 16 + ((lane & 16) ? 8 : 0);
#pragma unroll
  for (int v = 0; v < 8; ++v) {
    float val = acc[v] + bv;
    int idx = (rbase + v) * N + col;
    if (OUT16) C16[idx] = (_Float16)val;
    else       C32[idx] = val;
  }
}

// ---- persistent bidirectional GRU scan ------------------------------------
// 2 blocks (dir), 16 waves. h state in LDS; Wh B-fragments become resident
// VGPRs (loop-invariant). Wave w owns r-tile (mt=w&1, nt=w>>1), z-tile
// (nt=8+(w>>1)) and n-tile (nt=w>>1); h A-frags loaded once per step and
// shared by the r and z WMMA chains.

__global__ void __launch_bounds__(512)
gru_scan_kernel(const float* __restrict__ gi_f, const float* __restrict__ gi_b,
                const _Float16* __restrict__ whf, const _Float16* __restrict__ whb,
                const float* __restrict__ bhf, const float* __restrict__ bhb,
                _Float16* __restrict__ hc) {
  const int dir = blockIdx.x;
  const float*    gi   = dir ? gi_b : gi_f;
  const _Float16* WhRZ = dir ? whb : whf;      // rows [0,256): r,z
  const _Float16* WhN  = WhRZ + 2 * cH * cH;   // rows [256,384): n
  const float*    bh   = dir ? bhb : bhf;

  __shared__ __align__(16) _Float16 h16[cB * cH];
  __shared__ __align__(16) _Float16 rh16[cB * cH];
  __shared__ float hf32[cB * cH];
  __shared__ float zbuf[cB * cH];

  for (int i = threadIdx.x; i < cB * cH; i += blockDim.x) { h16[i] = (_Float16)0.f; hf32[i] = 0.f; }
  __syncthreads();

  const int wave = threadIdx.x >> 5;
  const int lane = threadIdx.x & 31;
  const int l15 = lane & 15;
  const int voff = (lane & 16) ? 8 : 0;
  const int m0 = (wave & 1) * 16;
  const int ntR = wave >> 1;           // 0..7  -> r gates / n gates
  const int ntZ = 8 + (wave >> 1);     // 8..15 -> z gates
  const int jR = ntR * 16 + l15;       // 0..127
  const int jZ = ntZ * 16 + l15;       // 128..255
  const float bjR = bh[jR];
  const float bjZ = bh[jZ];
  const float bjN = bh[2 * cH + jR];

  // loop-invariant weight B-fragments (compiler keeps these in VGPRs)
  v16h bR[4], bZ[4], bN[4];
#pragma unroll
  for (int kk = 0; kk < 4; ++kk) {
    bR[kk] = frag_b(WhRZ, cH, ntR * 16, kk * 32);
    bZ[kk] = frag_b(WhRZ, cH, ntZ * 16, kk * 32);
    bN[kk] = frag_b(WhN,  cH, ntR * 16, kk * 32);
  }

  for (int t = 0; t < cT; ++t) {
    const int tr = dir ? (cT - 1 - t) : t;
    const float* giRow = gi + tr * cB * (3 * cH);

    // prefetch next step's gate-input slab (49KB) while WMMAs run
    if (t + 1 < cT) {
      const int trn = dir ? (cT - 2 - t) : (t + 1);
      __builtin_prefetch(gi + trn * cB * (3 * cH) + threadIdx.x * 32, 0, 1);
    }

    // ---- phase 1: r and z gates share the h A-fragments -----------------
    {
      v16h af[4];
#pragma unroll
      for (int kk = 0; kk < 4; ++kk) af[kk] = frag_a(h16, cH, m0, kk * 32);
      v8f accR = {}, accZ = {};
#pragma unroll
      for (int kk = 0; kk < 4; ++kk) accR = WMMA_F16(af[kk], bR[kk], accR);
#pragma unroll
      for (int kk = 0; kk < 4; ++kk) accZ = WMMA_F16(af[kk], bZ[kk], accZ);

      // batched gather of gi / h operands, then the transcendental chains
      float gR[8], gZ[8], hv[8];
#pragma unroll
      for (int v = 0; v < 8; ++v) {
        const int bb = m0 + voff + v;
        gR[v] = giRow[bb * (3 * cH) + jR];
        gZ[v] = giRow[bb * (3 * cH) + jZ];
        hv[v] = hf32[bb * cH + jR];
      }
#pragma unroll
      for (int v = 0; v < 8; ++v) {
        const int bb = m0 + voff + v;
        float r = fast_sigmoid(accR[v] + bjR + gR[v]);
        rh16[bb * cH + jR] = (_Float16)(r * hv[v]);
        float z = fast_sigmoid(accZ[v] + bjZ + gZ[v]);
        zbuf[bb * cH + (jZ - cH)] = z;
      }
    }
    __syncthreads();

    // ---- phase 2: hn = (r*h) @ WhN^T, combine, update state -------------
    {
      v8f acc = {};
#pragma unroll
      for (int kk = 0; kk < 4; ++kk) {
        v16h a = frag_a(rh16, cH, m0, kk * 32);
        acc = WMMA_F16(a, bN[kk], acc);
      }
      float gN[8], zv[8], hv[8];
#pragma unroll
      for (int v = 0; v < 8; ++v) {
        const int bb = m0 + voff + v;
        gN[v] = giRow[bb * (3 * cH) + 2 * cH + jR];
        zv[v] = zbuf[bb * cH + jR];
        hv[v] = hf32[bb * cH + jR];
      }
#pragma unroll
      for (int v = 0; v < 8; ++v) {
        const int bb = m0 + voff + v;
        float n = fast_tanh(acc[v] + bjN + gN[v]);
        float hn = (1.f - zv[v]) * n + zv[v] * hv[v];
        hf32[bb * cH + jR] = hn;
        h16[bb * cH + jR] = (_Float16)hn;
        hc[(tr * cB + bb) * (2 * cH) + dir * cH + jR] = (_Float16)hn;  // concat(hf,hb)
      }
    }
    __syncthreads();
  }
}

// ---- flash attention, one block per (batch, head), 8 waves ---------------

__global__ void __launch_bounds__(256)
attn_kernel(const _Float16* __restrict__ qkv, const _Float16* __restrict__ vT,
            _Float16* __restrict__ o) {
  const int bh = blockIdx.x;
  const int b = bh >> 3, hh = bh & 7;
  const int wave = threadIdx.x >> 5;
  const int lane = threadIdx.x & 31;
  const int l15 = lane & 15;
  const int voff = (lane & 16) ? 8 : 0;
  const int kbA = (lane & 16) ? 8 : 0;
  const int kbB = (lane & 16) ? 16 : 0;
  const float scale = 0.17677669529663687f;  // 1/sqrt(hd)

  __shared__ __align__(16) _Float16 stage[8][16 * 32];  // P tile C->A relayout, 1KB/wave
  _Float16* st = stage[wave];

  for (int qi = 0; qi < 4; ++qi) {
    const int qt = wave + qi * 8;   // 32 query tiles / 8 waves
    // Q A-fragment: row t = qt*16 + (l&15) -> memory row (t*B + b), cols hh*32..
    v16h aq = frag_from_split(qkv + ((qt * 16 + l15) * cB + b) * (3 * cE) + hh * cHD + kbA);

    float m[8], lsum[8];
    v8f o0 = {}, o1 = {};
#pragma unroll
    for (int v = 0; v < 8; ++v) { m[v] = -1e30f; lsum[v] = 0.f; }

    for (int kt2 = 0; kt2 < 16; ++kt2) {  // 32 keys per iteration
      // S = Q K^T for two 16-key tiles (K-dim = hd = 32 -> one WMMA each)
      v16h bk0 = frag_from_split(qkv + (((2 * kt2) * 16 + l15) * cB + b) * (3 * cE) + cE + hh * cHD + kbA);
      v16h bk1 = frag_from_split(qkv + (((2 * kt2 + 1) * 16 + l15) * cB + b) * (3 * cE) + cE + hh * cHD + kbA);
      v8f s0 = {}, s1 = {};
      s0 = WMMA_F16(aq, bk0, s0);
      s1 = WMMA_F16(aq, bk1, s1);

      // online softmax over these 32 columns; row reductions across 16-lane N groups
      float tm[8];
#pragma unroll
      for (int v = 0; v < 8; ++v) tm[v] = fmaxf(s0[v] * scale, s1[v] * scale);
#pragma unroll
      for (int off = 1; off < 16; off <<= 1)
#pragma unroll
        for (int v = 0; v < 8; ++v) tm[v] = fmaxf(tm[v], __shfl_xor(tm[v], off, 32));

      float sc[8], pr0[8], pr1[8], rs[8];
#pragma unroll
      for (int v = 0; v < 8; ++v) {
        float mn = fmaxf(m[v], tm[v]);
        sc[v] = __expf(m[v] - mn);
        pr0[v] = __expf(s0[v] * scale - mn);
        pr1[v] = __expf(s1[v] * scale - mn);
        m[v] = mn;
        rs[v] = pr0[v] + pr1[v];
      }
#pragma unroll
      for (int off = 1; off < 16; off <<= 1)
#pragma unroll
        for (int v = 0; v < 8; ++v) rs[v] += __shfl_xor(rs[v], off, 32);
#pragma unroll
      for (int v = 0; v < 8; ++v) lsum[v] = lsum[v] * sc[v] + rs[v];

      // C-layout -> A-layout for P through LDS stage  [16 rows][32 keys]
#pragma unroll
      for (int v = 0; v < 8; ++v) {
        st[(voff + v) * 32 + l15] = (_Float16)pr0[v];
        st[(voff + v) * 32 + 16 + l15] = (_Float16)pr1[v];
      }
      __syncthreads();
      v16h ap = frag_a(st, 32, 0, 0);

      // O += P @ V  (V^T rows contiguous over keys)
      v16h bv0 = frag_from_split(vT + (bh * cHD + l15) * cT + kt2 * 32 + kbA);
      v16h bv1 = frag_from_split(vT + (bh * cHD + 16 + l15) * cT + kt2 * 32 + kbA);
#pragma unroll
      for (int v = 0; v < 8; ++v) { o0[v] *= sc[v]; o1[v] *= sc[v]; }
      o0 = WMMA_F16(ap, bv0, o0);
      o1 = WMMA_F16(ap, bv1, o1);
      __syncthreads();
    }

#pragma unroll
    for (int v = 0; v < 8; ++v) {
      const int trow = qt * 16 + voff + v;
      const float inv = fast_rcp(lsum[v]);
      o[(trow * cB + b) * cE + hh * cHD + l15] = (_Float16)(o0[v] * inv);
      o[(trow * cB + b) * cE + hh * cHD + 16 + l15] = (_Float16)(o1[v] * inv);
    }
  }
}

// ---- host orchestration ---------------------------------------------------

extern "C" void kernel_launch(void* const* d_in, const int* in_sizes, int n_in,
                              void* d_out, int out_size, void* d_ws, size_t ws_size,
                              hipStream_t stream) {
  (void)in_sizes; (void)n_in; (void)out_size; (void)ws_size;
  const float* x         = (const float*)d_in[0];
  const float* mask      = (const float*)d_in[1];
  const float* fwd_Wi    = (const float*)d_in[2];
  const float* fwd_bi    = (const float*)d_in[3];
  const float* fwd_Wh    = (const float*)d_in[4];
  const float* fwd_bh    = (const float*)d_in[5];
  const float* bwd_Wi    = (const float*)d_in[6];
  const float* bwd_bi    = (const float*)d_in[7];
  const float* bwd_Wh    = (const float*)d_in[8];
  const float* bwd_bh    = (const float*)d_in[9];
  const float* attn_w_in = (const float*)d_in[10];
  const float* attn_b_in = (const float*)d_in[11];
  const float* attn_w_out= (const float*)d_in[12];
  const float* attn_b_out= (const float*)d_in[13];
  const float* out_w     = (const float*)d_in[14];
  const float* out_b     = (const float*)d_in[15];

  char* wsb = (char*)d_ws;
  size_t off = 0;
  auto alloc = [&](size_t bytes) -> void* {
    void* p = wsb + off;
    off = (off + bytes + 255) & ~(size_t)255;
    return p;
  };
  _Float16* xm_h   = (_Float16*)alloc((size_t)cTB * cD * 2);
  _Float16* wi_f   = (_Float16*)alloc((size_t)3 * cH * cD * 2);
  _Float16* wi_b   = (_Float16*)alloc((size_t)3 * cH * cD * 2);
  _Float16* wh_f   = (_Float16*)alloc((size_t)3 * cH * cH * 2);
  _Float16* wh_b   = (_Float16*)alloc((size_t)3 * cH * cH * 2);
  _Float16* win_h  = (_Float16*)alloc((size_t)3 * cE * cE * 2);
  _Float16* wout_h = (_Float16*)alloc((size_t)cE * cE * 2);
  _Float16* ow_h   = (_Float16*)alloc((size_t)cD * cE * 2);
  float*    gi_f   = (float*)alloc((size_t)cTB * 3 * cH * 4);
  float*    gi_b   = (float*)alloc((size_t)cTB * 3 * cH * 4);
  _Float16* hc_h   = (_Float16*)alloc((size_t)cTB * 2 * cH * 2);
  _Float16* qkv_h  = (_Float16*)alloc((size_t)cTB * 3 * cE * 2);
  _Float16* vT_h   = (_Float16*)alloc((size_t)cB * cNH * cHD * cT * 2);
  _Float16* o_h    = (_Float16*)alloc((size_t)cTB * cE * 2);
  _Float16* proj_h = (_Float16*)alloc((size_t)cTB * cE * 2);
  float*    imp    = (float*)alloc((size_t)cTB * cD * 4);

  const int TPB = 256;
  k_prep_xm<<<(cTB * cD + TPB - 1) / TPB, TPB, 0, stream>>>(x, mask, xm_h);
  k_cvt<<<(3 * cH * cD + TPB - 1) / TPB, TPB, 0, stream>>>(fwd_Wi, wi_f, 3 * cH * cD);
  k_cvt<<<(3 * cH * cD + TPB - 1) / TPB, TPB, 0, stream>>>(bwd_Wi, wi_b, 3 * cH * cD);
  k_cvt<<<(3 * cH * cH + TPB - 1) / TPB, TPB, 0, stream>>>(fwd_Wh, wh_f, 3 * cH * cH);
  k_cvt<<<(3 * cH * cH + TPB - 1) / TPB, TPB, 0, stream>>>(bwd_Wh, wh_b, 3 * cH * cH);
  k_cvt<<<(3 * cE * cE + TPB - 1) / TPB, TPB, 0, stream>>>(attn_w_in, win_h, 3 * cE * cE);
  k_cvt<<<(cE * cE + TPB - 1) / TPB, TPB, 0, stream>>>(attn_w_out, wout_h, cE * cE);
  k_cvt<<<(cD * cE + TPB - 1) / TPB, TPB, 0, stream>>>(out_w, ow_h, cD * cE);

  // gi = xm @ Wi^T + bi   (hoisted out of the recurrent scan)
  dim3 g1(cTB / 16, (3 * cH) / 64);
  gemm_xwT<cD, false><<<g1, 128, 0, stream>>>(xm_h, wi_f, fwd_bi, gi_f, nullptr, 3 * cH);
  gemm_xwT<cD, false><<<g1, 128, 0, stream>>>(xm_h, wi_b, bwd_bi, gi_b, nullptr, 3 * cH);

  gru_scan_kernel<<<2, 512, 0, stream>>>(gi_f, gi_b, wh_f, wh_b, fwd_bh, bwd_bh, hc_h);

  // qkv = hc @ Win^T + bin
  dim3 g2(cTB / 16, (3 * cE) / 64);
  gemm_xwT<2 * cH, true><<<g2, 128, 0, stream>>>(hc_h, win_h, attn_b_in, nullptr, qkv_h, 3 * cE);

  k_transpose_v<<<(cB * cNH * cHD * cT + TPB - 1) / TPB, TPB, 0, stream>>>(qkv_h, vT_h);
  attn_kernel<<<cB * cNH, 256, 0, stream>>>(qkv_h, vT_h, o_h);

  // out-proj then final projection to D
  dim3 g3(cTB / 16, cE / 64);
  gemm_xwT<cE, true><<<g3, 128, 0, stream>>>(o_h, wout_h, attn_b_out, nullptr, proj_h, cE);
  dim3 g4(cTB / 16, cD / 64);
  gemm_xwT<cE, false><<<g4, 128, 0, stream>>>(proj_h, ow_h, out_b, imp, nullptr, cD);

  k_final<<<(cB * cT * cD + TPB - 1) / TPB, TPB, 0, stream>>>(x, mask, imp, (float*)d_out);
}
